// BiMP_90950227460158
// MI455X (gfx1250) — matrix operand order repaired and checked by reading, not compiled, not run on hardware
//
#include <hip/hip_runtime.h>
#include <hip/hip_bf16.h>
#include <math.h>

#define NSN 4096
#define NTN 2048
#define NH  4
#define NC  32
#define DD  128
#define NE  131072
#define NCHUNK 8
#define SCH (NSN / NCHUNK)

typedef __bf16 bf16_t;
typedef bf16_t v16bf __attribute__((ext_vector_type(16)));
typedef bf16_t v8bf  __attribute__((ext_vector_type(8)));
typedef float  v8f   __attribute__((ext_vector_type(8)));

// ---------------- float atomic min/max via int bit tricks ----------------
__device__ __forceinline__ void atomicMaxFloat(float* addr, float val) {
    if (val >= 0.f) atomicMax((int*)addr, __float_as_int(val));
    else            atomicMin((unsigned int*)addr, __float_as_uint(val));
}
__device__ __forceinline__ void atomicMinFloat(float* addr, float val) {
    if (val >= 0.f) atomicMin((int*)addr, __float_as_int(val));
    else            atomicMax((unsigned int*)addr, __float_as_uint(val));
}

// ---------------- conversion / setup kernels ----------------
__global__ void k_cvt_bf16(bf16_t* __restrict__ dst, const float* __restrict__ src, size_t n) {
    size_t i = (size_t)blockIdx.x * blockDim.x + threadIdx.x;
    if (i < n) dst[i] = (bf16_t)src[i];
}

// dst[c*R + r] = bf16(src[r*C + c])  (produce N-major, K-contiguous weights)
__global__ void k_transpose_bf16(bf16_t* __restrict__ dst, const float* __restrict__ src, int R, int C) {
    int i = blockIdx.x * blockDim.x + threadIdx.x;
    if (i >= R * C) return;
    int r = i / C, c = i % C;
    dst[(size_t)c * R + r] = (bf16_t)src[i];
}

__global__ void k_fill(float* __restrict__ p, float v, int n) {
    int i = blockIdx.x * blockDim.x + threadIdx.x;
    if (i < n) p[i] = v;
}

// ---------------- WMMA bf16 GEMM: C[MxN] = A[MxK] * BT[NxK]^T + bias ----------------
// wave32; A fragment layout (ISA 7.12.2 16-bit A 16x32):
//   lanes 0-15  : row M=lane,    K = {k0..k0+7}  and {k0+16..k0+23}
//   lanes 16-31 : row M=lane-16, K = {k0+8..k0+15} and {k0+24..k0+31}
__device__ __forceinline__ v16bf load_a_frag(const bf16_t* A, int K, int m0, int k0, int lane) {
    int m  = m0 + (lane & 15);
    int kb = k0 + ((lane >> 4) << 3);           // +0 or +8
    const bf16_t* p = A + (size_t)m * K + kb;
    union { v16bf v; v8bf h[2]; } u;
    u.h[0] = *(const v8bf*)p;                   // K = kb .. kb+7
    u.h[1] = *(const v8bf*)(p + 16);            // K = kb+16 .. kb+23
    return u.v;
}
// B fragment (32x16, 16-bit): lanes 0-15 hold K={k0..k0+7} and {k0+16..k0+23} of column N=lane,
// lanes 16-31 hold K={k0+8..k0+15} and {k0+24..k0+31} of column N=lane-16. BT is NxK row-major.
__device__ __forceinline__ v16bf load_b_frag(const bf16_t* BT, int K, int n, int k0, int lane) {
    int kb = k0 + ((lane >> 4) << 3);           // +0 or +8
    const bf16_t* p = BT + (size_t)n * K + kb;
    union { v16bf v; v8bf h[2]; } u;
    u.h[0] = *(const v8bf*)p;
    u.h[1] = *(const v8bf*)(p + 16);
    return u.v;
}

__global__ void k_gemm(const bf16_t* __restrict__ A, const bf16_t* __restrict__ BT,
                       const float* __restrict__ bias, float* __restrict__ C,
                       int M, int N, int K) {
    int lane = threadIdx.x & 31;
    int wave = threadIdx.x >> 5;                   // 0..3
    int m0 = (blockIdx.x * 4 + wave) * 16;
    int n0 = blockIdx.y * 64;
    v8f acc[4] = {v8f{}, v8f{}, v8f{}, v8f{}};
    for (int k0 = 0; k0 < K; k0 += 32) {
        v16bf a = load_a_frag(A, K, m0, k0, lane);
#pragma unroll
        for (int nt = 0; nt < 4; ++nt) {
            v16bf b = load_b_frag(BT, K, n0 + nt * 16 + (lane & 15), k0, lane);
            acc[nt] = __builtin_amdgcn_wmma_f32_16x16x32_bf16(
                false, a, false, b, (short)0, acc[nt], false, false);
        }
    }
    // C/D 16x16 f32 layout: VGPR j <-> M = j (lanes 0-15) or j+8 (lanes 16-31); N = lane&15
#pragma unroll
    for (int nt = 0; nt < 4; ++nt) {
        int n = n0 + nt * 16 + (lane & 15);
        float bv = bias ? bias[n] : 0.0f;
#pragma unroll
        for (int j = 0; j < 8; ++j) {
            int row = m0 + j + ((lane & 16) ? 8 : 0);
            C[(size_t)row * N + n] = acc[nt][j] + bv;
        }
    }
}

// ---------------- sparse edge attention (layer 1) ----------------
// one wave per edge; lanes = channels within each head
__global__ void k_edge_alpha(const int* __restrict__ src, const int* __restrict__ dst,
                             const float* __restrict__ attr,
                             const float* __restrict__ q1, const float* __restrict__ k1,
                             const float* __restrict__ e1w,
                             float* __restrict__ alpha, float* __restrict__ amax) {
    int gw = (blockIdx.x * blockDim.x + threadIdx.x) >> 5;
    int lane = threadIdx.x & 31;
    if (gw >= NE) return;
    int s = src[gw], d = dst[gw];
    float a = attr[gw];
#pragma unroll
    for (int h = 0; h < NH; ++h) {
        int c = h * NC + lane;
        float p = q1[(size_t)d * DD + c] * (k1[(size_t)s * DD + c] + a * e1w[c]);
#pragma unroll
        for (int off = 16; off > 0; off >>= 1) p += __shfl_xor(p, off, 32);
        p *= 0.17677669529663687f;   // 1/sqrt(32)
        if (lane == 0) {
            alpha[gw * NH + h] = p;
            atomicMaxFloat(&amax[d * NH + h], p);
        }
    }
}

__global__ void k_amax_fix(float* __restrict__ amax, int n) {
    int i = blockIdx.x * blockDim.x + threadIdx.x;
    if (i < n && isinf(amax[i])) amax[i] = 0.0f;
}

__global__ void k_edge_exp(const int* __restrict__ dst, float* __restrict__ alpha,
                           const float* __restrict__ amax, float* __restrict__ denom) {
    int i = blockIdx.x * blockDim.x + threadIdx.x;      // < NE*NH
    if (i >= NE * NH) return;
    int e = i >> 2, h = i & 3;
    int d = dst[e];
    float ex = __expf(alpha[i] - amax[d * NH + h]);
    alpha[i] = ex;
    atomicAdd(&denom[d * NH + h], ex);
}

__global__ void k_edge_agg(const int* __restrict__ src, const int* __restrict__ dst,
                           const float* __restrict__ attr,
                           const float* __restrict__ v1, const float* __restrict__ e1w,
                           const float* __restrict__ ex, const float* __restrict__ denom,
                           float* __restrict__ agg) {
    size_t i = (size_t)blockIdx.x * blockDim.x + threadIdx.x;  // < NE*DD
    if (i >= (size_t)NE * DD) return;
    int e = (int)(i >> 7);
    int c = (int)(i & (DD - 1));
    int h = c >> 5;
    int d = dst[e], s = src[e];
    float attn = ex[e * NH + h] / (denom[d * NH + h] + 1e-16f);
    float val = (v1[(size_t)s * DD + c] + attr[e] * e1w[c]) * attn;
    atomicAdd(&agg[(size_t)d * DD + c], val);
}

// ---------------- GraphNorm (stats over axis 0, per column) ----------------
__global__ void k_gn_stats(const float* __restrict__ X, const float* __restrict__ X2, int R,
                           const float* __restrict__ ms,
                           float* __restrict__ mean, float* __restrict__ invstd) {
    int c = blockIdx.x, tid = threadIdx.x;
    float s = 0.f, s2 = 0.f;
    for (int r = tid; r < R; r += blockDim.x) {
        float v = X[(size_t)r * DD + c];
        if (X2) v += X2[(size_t)r * DD + c];
        s += v; s2 += v * v;
    }
    __shared__ float sh[256], sh2[256];
    sh[tid] = s; sh2[tid] = s2; __syncthreads();
    for (int st = 128; st > 0; st >>= 1) {
        if (tid < st) { sh[tid] += sh[tid + st]; sh2[tid] += sh2[tid + st]; }
        __syncthreads();
    }
    if (tid == 0) {
        float m = sh[0] / R;
        float msv = ms[c];
        float var = sh2[0] / R - 2.f * msv * m * m + msv * msv * m * m;  // E[(x-mean*ms)^2]
        mean[c] = m;
        invstd[c] = rsqrtf(var + 1e-5f);
    }
}

__global__ void k_gn1_apply(const float* __restrict__ agg, const float* __restrict__ s1,
                            const float* __restrict__ wv, const float* __restrict__ bv,
                            const float* __restrict__ ms, const float* __restrict__ mean,
                            const float* __restrict__ invstd, bf16_t* __restrict__ dst) {
    int i = blockIdx.x * blockDim.x + threadIdx.x;      // < NSN*DD
    int c = i & (DD - 1);
    float xv = agg[i] + s1[i];
    float y = wv[c] * (xv - mean[c] * ms[c]) * invstd[c] + bv[c];
    dst[i] = (bf16_t)fmaxf(y, 0.f);
}

__global__ void k_gn2_apply(const float* __restrict__ outbip,
                            const float* __restrict__ wv, const float* __restrict__ bv,
                            const float* __restrict__ ms, const float* __restrict__ mean,
                            const float* __restrict__ invstd, bf16_t* __restrict__ xtb) {
    int i = blockIdx.x * blockDim.x + threadIdx.x;      // < NTN*DD
    int c = i & (DD - 1);
    float xv = outbip[(size_t)NSN * DD + i];
    float y = wv[c] * (xv - mean[c] * ms[c]) * invstd[c] + bv[c];
    xtb[i] = (bf16_t)fmaxf(y, 0.f);
}

// ---------------- dense bipartite attention (flash, chunked) ----------------
__global__ void k_flash_partial(const float* __restrict__ q2, const float* __restrict__ k2,
                                const float* __restrict__ v2,
                                float* __restrict__ pm, float* __restrict__ pl,
                                float* __restrict__ pacc) {
    int gw = (blockIdx.x * blockDim.x + threadIdx.x) >> 5;  // (t, h, chunk)
    int lane = threadIdx.x & 31;
    if (gw >= NTN * NH * NCHUNK) return;
    int ch = gw & (NCHUNK - 1);
    int h  = (gw >> 3) & (NH - 1);
    int t  = gw >> 5;
    const float scale = 0.17677669529663687f;               // 1/sqrt(32)
    float qv = q2[(size_t)t * DD + h * NC + lane] * scale;
    float m = -INFINITY, l = 0.f, acc = 0.f;
    int sbase = ch * SCH;
    for (int i = 0; i < SCH; ++i) {
        int s = sbase + i;
        float p = qv * k2[(size_t)s * DD + h * NC + lane];
#pragma unroll
        for (int off = 16; off > 0; off >>= 1) p += __shfl_xor(p, off, 32);
        float mn = fmaxf(m, p);
        float cf = __expf(m - mn);
        float pe = __expf(p - mn);
        l = l * cf + pe;
        acc = acc * cf + pe * v2[(size_t)s * DD + h * NC + lane];
        m = mn;
    }
    if (lane == 0) { pm[gw] = m; pl[gw] = l; }
    pacc[(size_t)gw * 32 + lane] = acc;
}

__global__ void k_flash_combine(const float* __restrict__ pm, const float* __restrict__ pl,
                                const float* __restrict__ pacc, float* __restrict__ outbip) {
    int gw = (blockIdx.x * blockDim.x + threadIdx.x) >> 5;  // (t, h)
    int lane = threadIdx.x & 31;
    if (gw >= NTN * NH) return;
    int t = gw >> 2, h = gw & 3;
    int base = gw * NCHUNK;
    float M = -INFINITY;
#pragma unroll
    for (int ch = 0; ch < NCHUNK; ++ch) M = fmaxf(M, pm[base + ch]);
    float L = 0.f, A = 0.f;
#pragma unroll
    for (int ch = 0; ch < NCHUNK; ++ch) {
        float w = __expf(pm[base + ch] - M);
        L += w * pl[base + ch];
        A += w * pacc[(size_t)(base + ch) * 32 + lane];
    }
    outbip[(size_t)(NSN + t) * DD + h * NC + lane] += A / L;
}

// ---------------- output min/max normalize ----------------
__global__ void k_init_mm(float* mm) {
    if (threadIdx.x == 0 && blockIdx.x == 0) { mm[0] = INFINITY; mm[1] = -INFINITY; }
}
__global__ void k_minmax(const float* __restrict__ X, int n, float* __restrict__ mm) {
    int stride = gridDim.x * blockDim.x;
    float lmin = INFINITY, lmax = -INFINITY;
    for (int i = blockIdx.x * blockDim.x + threadIdx.x; i < n; i += stride) {
        float v = X[i];
        lmin = fminf(lmin, v); lmax = fmaxf(lmax, v);
    }
#pragma unroll
    for (int off = 16; off > 0; off >>= 1) {
        lmin = fminf(lmin, __shfl_xor(lmin, off, 32));
        lmax = fmaxf(lmax, __shfl_xor(lmax, off, 32));
    }
    if ((threadIdx.x & 31) == 0) { atomicMinFloat(&mm[0], lmin); atomicMaxFloat(&mm[1], lmax); }
}
__global__ void k_finalize(float* __restrict__ out, const float* __restrict__ mm, int n) {
    int i = blockIdx.x * blockDim.x + threadIdx.x;
    if (i < n) out[i] = (out[i] - mm[0]) / (mm[1] - mm[0] + 1e-8f);
}

// ---------------- host orchestration ----------------
extern "C" void kernel_launch(void* const* d_in, const int* in_sizes, int n_in,
                              void* d_out, int out_size, void* d_ws, size_t ws_size,
                              hipStream_t stream) {
    (void)in_sizes; (void)n_in; (void)out_size; (void)ws_size;
    const float* x     = (const float*)d_in[0];
    const int*   pei   = (const int*)d_in[1];      // [2, NE]
    const float* eattr = (const float*)d_in[2];
    const float* tne   = (const float*)d_in[3];
    const float* q1_w  = (const float*)d_in[4];
    const float* q1_b  = (const float*)d_in[5];
    const float* k1_w  = (const float*)d_in[6];
    const float* k1_b  = (const float*)d_in[7];
    const float* v1_w  = (const float*)d_in[8];
    const float* v1_b  = (const float*)d_in[9];
    const float* e1_w  = (const float*)d_in[10];
    const float* s1_w  = (const float*)d_in[11];
    const float* s1_b  = (const float*)d_in[12];
    const float* gn1w  = (const float*)d_in[13];
    const float* gn1b  = (const float*)d_in[14];
    const float* gn1ms = (const float*)d_in[15];
    const float* q2_w  = (const float*)d_in[16];
    const float* q2_b  = (const float*)d_in[17];
    const float* k2_w  = (const float*)d_in[18];
    const float* k2_b  = (const float*)d_in[19];
    const float* v2_w  = (const float*)d_in[20];
    const float* v2_b  = (const float*)d_in[21];
    const float* s2_w  = (const float*)d_in[22];
    const float* s2_b  = (const float*)d_in[23];
    const float* gn2w  = (const float*)d_in[24];
    const float* gn2b  = (const float*)d_in[25];
    const float* gn2ms = (const float*)d_in[26];
    float* out = (float*)d_out;

    const int* esrc = pei;
    const int* edst = pei + NE;

    // workspace arena
    uintptr_t cur = (uintptr_t)d_ws;
    auto alloc = [&](size_t bytes) -> void* {
        void* p = (void*)cur;
        cur += (bytes + 255) & ~(size_t)255;
        return p;
    };
    bf16_t* xb    = (bf16_t*)alloc((size_t)NSN * NSN * 2);       // x in bf16 (32 MB, L2-resident)
    bf16_t* w1tq  = (bf16_t*)alloc((size_t)DD * NSN * 2);
    bf16_t* w1tk  = (bf16_t*)alloc((size_t)DD * NSN * 2);
    bf16_t* w1tv  = (bf16_t*)alloc((size_t)DD * NSN * 2);
    bf16_t* w1ts  = (bf16_t*)alloc((size_t)DD * NSN * 2);
    float*  q1f   = (float*)alloc((size_t)NSN * DD * 4);
    float*  k1f   = (float*)alloc((size_t)NSN * DD * 4);
    float*  v1f   = (float*)alloc((size_t)NSN * DD * 4);
    float*  s1f   = (float*)alloc((size_t)NSN * DD * 4);
    float*  alpha = (float*)alloc((size_t)NE * NH * 4);
    float*  amax  = (float*)alloc((size_t)NSN * NH * 4);
    float*  denom = (float*)alloc((size_t)NSN * NH * 4);
    float*  agg   = (float*)alloc((size_t)NSN * DD * 4);
    float*  mean1 = (float*)alloc(DD * 4);
    float*  inv1  = (float*)alloc(DD * 4);
    bf16_t* xbipb = (bf16_t*)alloc((size_t)(NSN + NTN) * DD * 2);
    bf16_t* w2tq  = (bf16_t*)alloc((size_t)DD * DD * 2);
    bf16_t* w2tk  = (bf16_t*)alloc((size_t)DD * DD * 2);
    bf16_t* w2tv  = (bf16_t*)alloc((size_t)DD * DD * 2);
    bf16_t* w2ts  = (bf16_t*)alloc((size_t)DD * DD * 2);
    float*  q2f   = (float*)alloc((size_t)NTN * DD * 4);
    float*  k2f   = (float*)alloc((size_t)NSN * DD * 4);
    float*  v2f   = (float*)alloc((size_t)NSN * DD * 4);
    float*  outbip= (float*)alloc((size_t)(NSN + NTN) * DD * 4);
    float*  pm    = (float*)alloc((size_t)NTN * NH * NCHUNK * 4);
    float*  pl    = (float*)alloc((size_t)NTN * NH * NCHUNK * 4);
    float*  pacc  = (float*)alloc((size_t)NTN * NH * NCHUNK * 32 * 4);
    float*  mean2 = (float*)alloc(DD * 4);
    float*  inv2  = (float*)alloc(DD * 4);
    bf16_t* xtb   = (bf16_t*)alloc((size_t)NTN * DD * 2);
    float*  mm    = (float*)alloc(2 * 4);

    // --- setup conversions ---
    k_cvt_bf16<<<(NSN * (size_t)NSN + 255) / 256, 256, 0, stream>>>(xb, x, (size_t)NSN * NSN);
    k_transpose_bf16<<<(NSN * DD + 255) / 256, 256, 0, stream>>>(w1tq, q1_w, NSN, DD);
    k_transpose_bf16<<<(NSN * DD + 255) / 256, 256, 0, stream>>>(w1tk, k1_w, NSN, DD);
    k_transpose_bf16<<<(NSN * DD + 255) / 256, 256, 0, stream>>>(w1tv, v1_w, NSN, DD);
    k_transpose_bf16<<<(NSN * DD + 255) / 256, 256, 0, stream>>>(w1ts, s1_w, NSN, DD);

    // --- layer-1 projections (WMMA bf16) ---
    dim3 g1(NSN / 64, DD / 64);
    k_gemm<<<g1, 128, 0, stream>>>(xb, w1tq, q1_b, q1f, NSN, DD, NSN);
    k_gemm<<<g1, 128, 0, stream>>>(xb, w1tk, k1_b, k1f, NSN, DD, NSN);
    k_gemm<<<g1, 128, 0, stream>>>(xb, w1tv, v1_b, v1f, NSN, DD, NSN);
    k_gemm<<<g1, 128, 0, stream>>>(xb, w1ts, s1_b, s1f, NSN, DD, NSN);

    // --- sparse edge attention ---
    k_fill<<<(NSN * NH + 255) / 256, 256, 0, stream>>>(amax, -INFINITY, NSN * NH);
    k_fill<<<(NSN * NH + 255) / 256, 256, 0, stream>>>(denom, 0.f, NSN * NH);
    k_fill<<<(NSN * DD + 255) / 256, 256, 0, stream>>>(agg, 0.f, NSN * DD);
    k_edge_alpha<<<NE * 32 / 256, 256, 0, stream>>>(esrc, edst, eattr, q1f, k1f, e1_w, alpha, amax);
    k_amax_fix<<<(NSN * NH + 255) / 256, 256, 0, stream>>>(amax, NSN * NH);
    k_edge_exp<<<(NE * NH + 255) / 256, 256, 0, stream>>>(edst, alpha, amax, denom);
    k_edge_agg<<<((size_t)NE * DD + 255) / 256, 256, 0, stream>>>(esrc, edst, eattr, v1f, e1_w, alpha, denom, agg);

    // --- graph norm 1 + relu -> bf16 x_bip rows [0, NSN) ---
    k_gn_stats<<<DD, 256, 0, stream>>>(agg, s1f, NSN, gn1ms, mean1, inv1);
    k_gn1_apply<<<(NSN * DD) / 256, 256, 0, stream>>>(agg, s1f, gn1w, gn1b, gn1ms, mean1, inv1, xbipb);
    k_cvt_bf16<<<((size_t)NTN * DD + 255) / 256, 256, 0, stream>>>(xbipb + (size_t)NSN * DD, tne, (size_t)NTN * DD);

    // --- layer-2 projections (WMMA bf16) ---
    k_transpose_bf16<<<(DD * DD + 255) / 256, 256, 0, stream>>>(w2tq, q2_w, DD, DD);
    k_transpose_bf16<<<(DD * DD + 255) / 256, 256, 0, stream>>>(w2tk, k2_w, DD, DD);
    k_transpose_bf16<<<(DD * DD + 255) / 256, 256, 0, stream>>>(w2tv, v2_w, DD, DD);
    k_transpose_bf16<<<(DD * DD + 255) / 256, 256, 0, stream>>>(w2ts, s2_w, DD, DD);
    k_gemm<<<dim3(NTN / 64, DD / 64), 128, 0, stream>>>(xbipb + (size_t)NSN * DD, w2tq, q2_b, q2f, NTN, DD, DD);
    k_gemm<<<dim3(NSN / 64, DD / 64), 128, 0, stream>>>(xbipb, w2tk, k2_b, k2f, NSN, DD, DD);
    k_gemm<<<dim3(NSN / 64, DD / 64), 128, 0, stream>>>(xbipb, w2tv, v2_b, v2f, NSN, DD, DD);
    k_gemm<<<dim3((NSN + NTN) / 64, DD / 64), 128, 0, stream>>>(xbipb, w2ts, s2_b, outbip, NSN + NTN, DD, DD);

    // --- dense bipartite attention (flash) ---
    k_flash_partial<<<NTN * NH * NCHUNK / 8, 256, 0, stream>>>(q2f, k2f, v2f, pm, pl, pacc);
    k_flash_combine<<<NTN * NH / 8, 256, 0, stream>>>(pm, pl, pacc, outbip);

    // --- graph norm 2 + relu -> bf16 xt ---
    k_gn_stats<<<DD, 256, 0, stream>>>(outbip, nullptr, NSN + NTN, gn2ms, mean2, inv2);
    k_gn2_apply<<<(NTN * DD) / 256, 256, 0, stream>>>(outbip, gn2w, gn2b, gn2ms, mean2, inv2, xtb);

    // --- adj = xt @ xt^T (WMMA bf16; BT layout == xt itself), min/max normalize in place ---
    k_gemm<<<dim3(NTN / 64, NTN / 64), 128, 0, stream>>>(xtb, xtb, nullptr, out, NTN, NTN, DD);
    k_init_mm<<<1, 32, 0, stream>>>(mm);
    k_minmax<<<1024, 256, 0, stream>>>(out, NTN * NTN, mm);
    k_finalize<<<(NTN * NTN + 255) / 256, 256, 0, stream>>>(out, mm, NTN * NTN);
}
